// DepthSepConv2d_82901458747584
// MI455X (gfx1250) — compile-verified, matrix operand order checked by reading, not run on hardware
//
#include <hip/hip_runtime.h>
#include <hip/hip_bf16.h>
#include <stdint.h>

typedef __attribute__((ext_vector_type(16))) __bf16 v16bf;
typedef __attribute__((ext_vector_type(8)))  float  v8f;
typedef __attribute__((ext_vector_type(4)))  unsigned int v4u;
typedef __attribute__((ext_vector_type(8)))  int    v8i;
typedef __attribute__((ext_vector_type(4)))  int    v4i;

#define CIN   256
#define COUT  512
#define NB    32
#define HH    56
#define WW    56
#define HWSZ  (HH * WW)          // 3136
#define NHW   (NB * HWSZ)        // 100352
#define BN_EPS 1e-5f

#if defined(__has_builtin)
#if __has_builtin(__builtin_amdgcn_tensor_load_to_lds)
#define HAVE_TDM 1
#endif
#endif
#ifndef HAVE_TDM
#define HAVE_TDM 0
#endif

static __device__ __forceinline__ uint16_t f32_to_bf16(float f) {
    union { float f; uint32_t u; } v; v.f = f;
    uint32_t u = v.u;
    return (uint16_t)((u + 0x7FFFu + ((u >> 16) & 1u)) >> 16);
}

// ---------------- Kernel 0: zero stats scratch ----------------
__global__ void zero_kernel(float* p, int n) {
    int i = blockIdx.x * 256 + threadIdx.x;
    if (i < n) p[i] = 0.0f;
}

// ---------------- Kernel 1: depthwise 3x3 (reflect pad) + bias + BN1 partial stats
__global__ void dwconv_kernel(const float* __restrict__ x,
                              const float* __restrict__ dw_w,
                              const float* __restrict__ dw_b,
                              float* __restrict__ y,
                              float* __restrict__ sum1,
                              float* __restrict__ sumsq1) {
    __shared__ float tile[58 * 58];
    __shared__ float rs[256];
    __shared__ float rq[256];
    const int c   = blockIdx.x & (CIN - 1);
    const int b   = blockIdx.x >> 8;
    const int tid = threadIdx.x;
    const float* xp = x + ((size_t)b * CIN + c) * HWSZ;

    for (int i = tid; i < 58 * 58; i += 256) {
        int hh = i / 58 - 1, ww = i % 58 - 1;
        int sh = (hh < 0) ? 1 : ((hh >= HH) ? HH - 2 : hh);
        int sw = (ww < 0) ? 1 : ((ww >= WW) ? WW - 2 : ww);
        tile[i] = xp[sh * WW + sw];
    }
    const float w0 = dw_w[c * 9 + 0], w1 = dw_w[c * 9 + 1], w2 = dw_w[c * 9 + 2];
    const float w3 = dw_w[c * 9 + 3], w4 = dw_w[c * 9 + 4], w5 = dw_w[c * 9 + 5];
    const float w6 = dw_w[c * 9 + 6], w7 = dw_w[c * 9 + 7], w8 = dw_w[c * 9 + 8];
    const float bias = dw_b[c];
    __syncthreads();

    float s = 0.f, q = 0.f;
    float* yp = y + ((size_t)b * CIN + c) * HWSZ;
    for (int i = tid; i < HWSZ; i += 256) {
        int h = i / WW, w = i - h * WW;
        const float* t = &tile[h * 58 + w];
        float acc = bias;
        acc += t[0]   * w0 + t[1]   * w1 + t[2]   * w2;
        acc += t[58]  * w3 + t[59]  * w4 + t[60]  * w5;
        acc += t[116] * w6 + t[117] * w7 + t[118] * w8;
        yp[i] = acc;
        s += acc; q += acc * acc;
    }
    rs[tid] = s; rq[tid] = q;
    __syncthreads();
    for (int off = 128; off > 0; off >>= 1) {
        if (tid < off) { rs[tid] += rs[tid + off]; rq[tid] += rq[tid + off]; }
        __syncthreads();
    }
    if (tid == 0) {
        atomicAdd(&sum1[c],   rs[0]);
        atomicAdd(&sumsq1[c], rq[0]);
    }
}

// ---------------- Kernel 2: fold BN1 into scale/shift ----------------
__global__ void bn1_params(const float* __restrict__ sum1, const float* __restrict__ sumsq1,
                           const float* __restrict__ g1, const float* __restrict__ b1,
                           float* __restrict__ scale1, float* __restrict__ shift1) {
    int c = blockIdx.x * 256 + threadIdx.x;
    if (c < CIN) {
        float inv = 1.0f / (float)NHW;
        float m   = sum1[c] * inv;
        float var = sumsq1[c] * inv - m * m;
        float sc  = g1[c] * rsqrtf(var + BN_EPS);
        scale1[c] = sc;
        shift1[c] = b1[c] - m * sc;
    }
}

// ---------------- Kernel 3: BN1 apply + ReLU + bf16 + transpose ----------------
// y f32 [b][c][s]  ->  ybf bf16 [b][s][c]   (64x64 LDS-tiled transpose)
__global__ void bn1_apply_kernel(const float* __restrict__ y,
                                 const float* __restrict__ scale1,
                                 const float* __restrict__ shift1,
                                 uint16_t* __restrict__ ybf) {
    __shared__ uint16_t sT[64][72];       // [c][s], padded
    const int bx = blockIdx.x;            // 0..1567
    const int b  = bx / 49;
    const int s0 = (bx - b * 49) * 64;
    const int c0 = blockIdx.y * 64;
    const int tid = threadIdx.x;
    const int ss = tid & 63, cg = tid >> 6;
    #pragma unroll
    for (int cc = cg; cc < 64; cc += 4) {
        int c = c0 + cc;
        float v = y[((size_t)b * CIN + c) * HWSZ + s0 + ss];
        v = v * scale1[c] + shift1[c];
        v = v > 0.f ? v : 0.f;
        sT[cc][ss] = f32_to_bf16(v);
    }
    __syncthreads();
    const int cc2 = tid & 63, sg = tid >> 6;
    #pragma unroll
    for (int sr = sg; sr < 64; sr += 4) {
        ybf[((size_t)b * HWSZ + s0 + sr) * CIN + c0 + cc2] = sT[cc2][sr];
    }
}

// ---------------- Kernel 4: pw_w f32 -> bf16 ----------------
__global__ void wcvt_kernel(const float* __restrict__ pw_w, uint16_t* __restrict__ wbf) {
    int i = blockIdx.x * 256 + threadIdx.x;
    if (i < COUT * CIN) wbf[i] = f32_to_bf16(pw_w[i]);
}

// ---------------- TDM 2D tile load: global -> LDS (row pad 16B per 512B) ----
#if HAVE_TDM
static __device__ __forceinline__ void tdm_load_2d(uint32_t lds_addr, const void* gaddr,
                                                   uint32_t rows, uint32_t row_elems,
                                                   uint32_t row_stride_elems) {
    uint64_t ga = (uint64_t)(uintptr_t)gaddr;
    v4u g0;
    g0[0] = 1u;                                   // count=1 (valid user descriptor)
    g0[1] = lds_addr;                             // LDS byte address
    g0[2] = (uint32_t)ga;                         // global_addr[31:0]
    g0[3] = (uint32_t)((ga >> 32) & 0x1FFFFFFu)   // global_addr[56:32]
          | (2u << 30);                           // type=2 ("image")
    const uint32_t data_size = 1;                 // 2-byte elements (bf16)
    const uint32_t pad_en = 1, pad_int = 6, pad_amt = 3; // pad 4 DW after every 128 DW
    v8i g1;
    g1[0] = (int)((data_size << 16) | (pad_en << 20) | (pad_int << 22) | (pad_amt << 25));
    g1[1] = (int)((row_elems & 0xFFFFu) << 16);                                // tensor_dim0 lo
    g1[2] = (int)(((row_elems >> 16) & 0xFFFFu) | ((rows & 0xFFFFu) << 16));   // dim0 hi | tensor_dim1 lo
    g1[3] = (int)(((rows >> 16) & 0xFFFFu) | ((row_elems & 0xFFFFu) << 16));   // dim1 hi | tile_dim0
    g1[4] = (int)(rows & 0xFFFFu);                                             // tile_dim1 (tile_dim2=0)
    g1[5] = (int)row_stride_elems;                                             // tensor_dim0_stride lo
    g1[6] = 0;
    g1[7] = 0;
    v4i g2 = {0, 0, 0, 0};
    v4i g3 = {0, 0, 0, 0};
#if __clang_major__ >= 23
    v8i g4 = {0, 0, 0, 0, 0, 0, 0, 0};
    __builtin_amdgcn_tensor_load_to_lds(g0, g1, g2, g3, g4, 0);
#else
    __builtin_amdgcn_tensor_load_to_lds(g0, g1, g2, g3, 0);
#endif
}
#endif

// ---------------- Kernel 5: pointwise GEMM via WMMA bf16, TDM-fed LDS -------
// Block tile: 64 out-channels x 32 spatial, K=256 fully staged in LDS.
// 8 waves, each owns one 16x16 f32 accumulator over 8 K-steps of 32.
// LDS rows: 256 bf16 (512B) + 16B pad -> 528B stride: conflict-free b128 frags.
#define SROWE 264   // LDS row stride in uint16 elements (528 bytes)
__global__ void __launch_bounds__(256)
pw_gemm_kernel(const uint16_t* __restrict__ ybf,
               const uint16_t* __restrict__ wbf,
               const float* __restrict__ pw_b,
               float* __restrict__ out) {
    __shared__ __align__(16) uint16_t sW[64 * SROWE];   // [o][k]  33792 B
    __shared__ __align__(16) uint16_t sY[32 * SROWE];   // [s][k]  16896 B

    const int bx    = blockIdx.x;            // 0..3135 (98 s-tiles x 32 images)
    const int b     = bx / 98;
    const int s0    = (bx - b * 98) * 32;
    const int obase = blockIdx.y * 64;
    const int tid   = threadIdx.x;
    const int lane  = tid & 31;
    const int wave  = tid >> 5;
    const int oq    = (wave & 3) * 16;       // wave's o sub-tile
    const int sq    = (wave >> 2) * 16;      // wave's s sub-tile

#if HAVE_TDM
    if (wave == 0) {
        tdm_load_2d((uint32_t)(uintptr_t)&sW[0],
                    wbf + (size_t)obase * CIN, 64, CIN, CIN);
        tdm_load_2d((uint32_t)(uintptr_t)&sY[0],
                    ybf + ((size_t)b * HWSZ + s0) * CIN, 32, CIN, CIN);
        __builtin_amdgcn_s_wait_tensorcnt(0);
    }
    __syncthreads();
#else
    for (int ch = tid; ch < 2048; ch += 256) {          // W: 64 rows x 32 chunks
        int row = ch >> 5, kc = (ch & 31) * 8;
        *(uint4*)&sW[row * SROWE + kc] =
            *(const uint4*)&wbf[(size_t)(obase + row) * CIN + kc];
    }
    for (int ch = tid; ch < 1024; ch += 256) {          // Y: 32 rows x 32 chunks
        int row = ch >> 5, kc = (ch & 31) * 8;
        *(uint4*)&sY[row * SROWE + kc] =
            *(const uint4*)&ybf[((size_t)b * HWSZ + s0 + row) * CIN + kc];
    }
    __syncthreads();
#endif

    const int half16 = (lane >> 4) * 16;     // K-half within 32-step
    const int mrow   = lane & 15;

    v8f acc = {};
    #pragma unroll
    for (int kk = 0; kk < 8; ++kk) {
        const int kof = kk * 32 + half16;
        union { uint4 u[2]; v16bf v; } a, bb;
        a.u[0]  = *(const uint4*)&sW[(oq + mrow) * SROWE + kof];
        a.u[1]  = *(const uint4*)&sW[(oq + mrow) * SROWE + kof + 8];
        bb.u[0] = *(const uint4*)&sY[(sq + mrow) * SROWE + kof];
        bb.u[1] = *(const uint4*)&sY[(sq + mrow) * SROWE + kof + 8];
        acc = __builtin_amdgcn_wmma_f32_16x16x32_bf16(false, a.v, false, bb.v,
                                                      (short)0, acc, false, false);
    }

    const int m8 = (lane >> 4) * 8, ncol = lane & 15;
    #pragma unroll
    for (int r = 0; r < 8; ++r) {
        int o = obase + oq + m8 + r;
        out[((size_t)b * COUT + o) * HWSZ + s0 + sq + ncol] = acc[r] + pw_b[o];
    }
}

// ---------------- Kernel 6: BN2 stats (one block per out-channel) ----------
__global__ void bn2_stats(const float* __restrict__ out,
                          const float* __restrict__ g2, const float* __restrict__ b2,
                          float* __restrict__ scale2, float* __restrict__ shift2) {
    __shared__ float rs[256];
    __shared__ float rq[256];
    const int o = blockIdx.x;
    const int tid = threadIdx.x;
    float s = 0.f, q = 0.f;
    for (int i = tid; i < NHW; i += 256) {
        int bb = i / HWSZ;
        int sp = i - bb * HWSZ;
        float v = out[((size_t)bb * COUT + o) * HWSZ + sp];
        s += v; q += v * v;
    }
    rs[tid] = s; rq[tid] = q;
    __syncthreads();
    for (int off = 128; off > 0; off >>= 1) {
        if (tid < off) { rs[tid] += rs[tid + off]; rq[tid] += rq[tid + off]; }
        __syncthreads();
    }
    if (tid == 0) {
        float inv = 1.0f / (float)NHW;
        float m   = rs[0] * inv;
        float var = rq[0] * inv - m * m;
        float sc  = g2[o] * rsqrtf(var + BN_EPS);
        scale2[o] = sc;
        shift2[o] = b2[o] - m * sc;
    }
}

// ---------------- Kernel 7: BN2 + ReLU in place ----------------
__global__ void bn2_apply(float* __restrict__ out,
                          const float* __restrict__ scale2,
                          const float* __restrict__ shift2, size_t n) {
    size_t i = (size_t)blockIdx.x * 256 + threadIdx.x;
    const size_t stride = (size_t)gridDim.x * 256;
    for (; i < n; i += stride) {
        int o = (int)((i / HWSZ) % COUT);
        float v = out[i] * scale2[o] + shift2[o];
        out[i] = v > 0.f ? v : 0.f;
    }
}

extern "C" void kernel_launch(void* const* d_in, const int* in_sizes, int n_in,
                              void* d_out, int out_size, void* d_ws, size_t ws_size,
                              hipStream_t stream) {
    const float* x    = (const float*)d_in[0];
    const float* dw_w = (const float*)d_in[1];
    const float* dw_b = (const float*)d_in[2];
    const float* g1   = (const float*)d_in[3];
    const float* b1   = (const float*)d_in[4];
    const float* pw_w = (const float*)d_in[5];
    const float* pw_b = (const float*)d_in[6];
    const float* g2   = (const float*)d_in[7];
    const float* b2   = (const float*)d_in[8];
    float* out = (float*)d_out;

    // workspace layout (256B-aligned)
    char* ws = (char*)d_ws;
    float*    y     = (float*)ws;                                   // 25,690,112 f32
    uint16_t* ybf   = (uint16_t*)(ws + 102760448);                  // 25,690,112 bf16
    uint16_t* wbf   = (uint16_t*)(ws + 154140672);                  // 131,072 bf16
    float*    stats = (float*)(ws + 154402816);
    float* sum1   = stats;
    float* sumsq1 = stats + 256;
    float* scale1 = stats + 512;
    float* shift1 = stats + 768;
    float* scale2 = stats + 1024;
    float* shift2 = stats + 1536;

    zero_kernel<<<2, 256, 0, stream>>>(sum1, 512);
    dwconv_kernel<<<NB * CIN, 256, 0, stream>>>(x, dw_w, dw_b, y, sum1, sumsq1);
    bn1_params<<<1, 256, 0, stream>>>(sum1, sumsq1, g1, b1, scale1, shift1);
    dim3 tgrid(1568, CIN / 64);
    bn1_apply_kernel<<<tgrid, 256, 0, stream>>>(y, scale1, shift1, ybf);
    wcvt_kernel<<<(COUT * CIN) / 256, 256, 0, stream>>>(pw_w, wbf);
    dim3 ggrid(3136, COUT / 64);
    pw_gemm_kernel<<<ggrid, 256, 0, stream>>>(ybf, wbf, pw_b, out);
    bn2_stats<<<COUT, 256, 0, stream>>>(out, g2, b2, scale2, shift2);
    bn2_apply<<<4096, 256, 0, stream>>>(out, scale2, shift2, (size_t)NB * COUT * HWSZ);
}
